// MultiHeadAttention_9740985827891
// MI455X (gfx1250) — matrix operand order repaired and checked by reading, not compiled, not run on hardware
//
#include <hip/hip_runtime.h>

// ---- MHA for MI455X (gfx1250): f16 WMMA projections + fused flash-attention ----
// B=2, S=2048, E=1024, H=16, d=64.
// Round 4: hoist fragment loads ahead of WMMA bursts (back-to-back WMMA issue),
//          2x2 wave-quadrant proj GEMM (less LDS re-read), async wait fallback.

typedef __attribute__((ext_vector_type(16))) _Float16 v16h;
typedef __attribute__((ext_vector_type(8)))  _Float16 v8h;
typedef __attribute__((ext_vector_type(4)))  _Float16 v4h;
typedef __attribute__((ext_vector_type(8)))  float    v8f;

#define E_DIM  1024
#define H_NUM  16
#define D_HEAD 64
#define S_LEN  2048
#define B_NUM  2

// --- CDNA5 async global->LDS path (load builtin proven present on this TC) ---
#if defined(__has_builtin)
#if __has_builtin(__builtin_amdgcn_global_load_async_to_lds_b128)
#define USE_ASYNC_COPY 1
#if __has_builtin(__builtin_amdgcn_s_wait_asynccnt)
#define WAIT_ASYNC0() __builtin_amdgcn_s_wait_asynccnt(0)
#else
#define WAIT_ASYNC0() asm volatile("s_wait_asynccnt 0x0" ::: "memory")
#endif
#endif
#endif
#ifndef USE_ASYNC_COPY
#define USE_ASYNC_COPY 0
#define WAIT_ASYNC0() ((void)0)
#endif

#if USE_ASYNC_COPY
typedef int v4i_vec __attribute__((vector_size(16)));
typedef __attribute__((address_space(1))) v4i_vec* g1_v4i;
typedef __attribute__((address_space(3))) v4i_vec* l3_v4i;
static __device__ __forceinline__ void async_copy_b128(const void* gsrc, void* ldst) {
    __builtin_amdgcn_global_load_async_to_lds_b128(
        (g1_v4i)gsrc, (l3_v4i)ldst, 0, 0);
}
#endif

static __device__ __forceinline__ v8f v8f_zero() {
    v8f z;
#pragma unroll
    for (int i = 0; i < 8; ++i) z[i] = 0.0f;
    return z;
}

static __device__ __forceinline__ v16h cat8(v8h lo, v8h hi) {
    v16h r;
#pragma unroll
    for (int i = 0; i < 8; ++i) { r[i] = lo[i]; r[i + 8] = hi[i]; }
    return r;
}

static __device__ __forceinline__ v8h ldsv8(const _Float16* p) {
    return *(const v8h*)p;   // 16B-aligned by construction -> ds_load_b128
}

// reduce across the 16-lane half of a wave32 (masks 1,2,4,8 stay within the half)
static __device__ __forceinline__ float rowmax16(float x) {
#pragma unroll
    for (int m = 8; m >= 1; m >>= 1) x = fmaxf(x, __shfl_xor(x, m, 32));
    return x;
}
static __device__ __forceinline__ float rowsum16(float x) {
#pragma unroll
    for (int m = 8; m >= 1; m >>= 1) x += __shfl_xor(x, m, 32);
    return x;
}

static __device__ __forceinline__ v8f wmma_f16(v16h a, v16h b, v8f c) {
    return __builtin_amdgcn_wmma_f32_16x16x32_f16(false, a, false, b, (short)0, c,
                                                  false, false);
}

// ---------------------------------------------------------------------------
// Projection: Y = X @ W + bias, X:[4096,1024] f32, W:[1024,1024] f32.
// Output f16, head-major [B, H, S, 64]. Block = 128 threads (4 waves),
// 64x64 output tile; each wave owns a 32x32 quadrant (2 A-frags x 2 B-frags).
// K-steps of 32; double-buffered LDS, register prefetch, ONE barrier per step.
// ---------------------------------------------------------------------------
__global__ __launch_bounds__(128)
void mha_proj_kernel(const float* __restrict__ x,
                     const float* __restrict__ w,
                     const float* __restrict__ bias,
                     _Float16*    __restrict__ dst)
{
    __shared__ __align__(16) _Float16 lds_x[2][64 * 32];    // [row][k]
    __shared__ __align__(16) _Float16 lds_wt[2][64 * 32];   // [col][k] (transposed)

    const int tid     = threadIdx.x;
    const int wave    = tid >> 5;
    const int lane    = tid & 31;
    const int halfsel = lane >> 4;
    const int lm      = lane & 15;
    const int wrq     = wave & 1;    // row quadrant (0/1 -> rows 0-31 / 32-63)
    const int wcq     = wave >> 1;   // col quadrant (0/1 -> cols 0-31 / 32-63)

    const int r0 = blockIdx.x * 64;   // row tile (over B*S = 4096)
    const int c0 = blockIdx.y * 64;   // col tile (over H*D = 1024)

    v8f acc[2][2];
#pragma unroll
    for (int i = 0; i < 2; ++i)
#pragma unroll
        for (int j = 0; j < 2; ++j) acc[i][j] = v8f_zero();

    float4 xr[4], wr[4];
    // prefetch k-slice 0 into registers
#pragma unroll
    for (int i = 0; i < 4; ++i) {
        int idx = tid + i * 128, row = idx >> 3, c4 = (idx & 7) << 2;
        xr[i] = *(const float4*)&x[(size_t)(r0 + row) * E_DIM + c4];
    }
#pragma unroll
    for (int i = 0; i < 4; ++i) {
        int idx = tid + i * 128, k = idx >> 4, c4 = (idx & 15) << 2;
        wr[i] = *(const float4*)&w[(size_t)k * E_DIM + c0 + c4];
    }

    for (int kk = 0; kk < E_DIM; kk += 32) {
        const int cur = (kk >> 5) & 1;

        // Convert + store staged registers into LDS buffer `cur`.
#pragma unroll
        for (int i = 0; i < 4; ++i) {
            int idx = tid + i * 128, row = idx >> 3, c4 = (idx & 7) << 2;
            v4h hx;
            hx[0] = (_Float16)xr[i].x; hx[1] = (_Float16)xr[i].y;
            hx[2] = (_Float16)xr[i].z; hx[3] = (_Float16)xr[i].w;
            *(v4h*)&lds_x[cur][row * 32 + c4] = hx;
        }
#pragma unroll
        for (int i = 0; i < 4; ++i) {
            int idx = tid + i * 128, k = idx >> 4, c4 = (idx & 15) << 2;
            lds_wt[cur][(c4 + 0) * 32 + k] = (_Float16)wr[i].x;
            lds_wt[cur][(c4 + 1) * 32 + k] = (_Float16)wr[i].y;
            lds_wt[cur][(c4 + 2) * 32 + k] = (_Float16)wr[i].z;
            lds_wt[cur][(c4 + 3) * 32 + k] = (_Float16)wr[i].w;
        }
        __syncthreads();   // the only barrier in the k-loop

        // Prefetch next k-slice; global loads overlap the WMMAs below.
        if (kk + 32 < E_DIM) {
#pragma unroll
            for (int i = 0; i < 4; ++i) {
                int idx = tid + i * 128, row = idx >> 3, c4 = (idx & 7) << 2;
                xr[i] = *(const float4*)&x[(size_t)(r0 + row) * E_DIM + kk + 32 + c4];
            }
#pragma unroll
            for (int i = 0; i < 4; ++i) {
                int idx = tid + i * 128, k = idx >> 4, c4 = (idx & 15) << 2;
                wr[i] = *(const float4*)&w[(size_t)(kk + 32 + k) * E_DIM + c0 + c4];
            }
        }

        // Load ALL fragments first, then issue the 4 WMMAs back-to-back.
        // A (16x32): row = lane%16, K(j) = j + halfsel*16
        // B (32x16): N = lane%16, K(j) = j + halfsel*16
        v16h afr[2], bfr[2];
        const int abase = halfsel * 16;
        const int bbase = halfsel * 16;
#pragma unroll
        for (int i = 0; i < 2; ++i) {
            const int arow = wrq * 32 + i * 16 + lm;
            afr[i] = cat8(ldsv8(&lds_x[cur][arow * 32 + abase]),
                          ldsv8(&lds_x[cur][arow * 32 + abase + 8]));
        }
#pragma unroll
        for (int j = 0; j < 2; ++j) {
            const int bc = wcq * 32 + j * 16 + lm;
            bfr[j] = cat8(ldsv8(&lds_wt[cur][bc * 32 + bbase]),
                          ldsv8(&lds_wt[cur][bc * 32 + bbase + 8]));
        }
#pragma unroll
        for (int i = 0; i < 2; ++i)
#pragma unroll
            for (int j = 0; j < 2; ++j)
                acc[i][j] = wmma_f16(afr[i], bfr[j], acc[i][j]);
    }

    // Epilogue: + bias, convert to f16, scatter into [B, H, S, 64].
#pragma unroll
    for (int ai = 0; ai < 2; ++ai)
#pragma unroll
    for (int bj = 0; bj < 2; ++bj) {
        const int col = c0 + wcq * 32 + bj * 16 + lm;
        const float bv = bias[col];
        const int hh = col >> 6, dd = col & 63;
#pragma unroll
        for (int i = 0; i < 8; ++i) {
            int rg = r0 + wrq * 32 + ai * 16 + i + halfsel * 8;
            int bb = rg >> 11, ss = rg & (S_LEN - 1);
            dst[(((size_t)(bb * H_NUM + hh) * S_LEN) + ss) * D_HEAD + dd] =
                (_Float16)(acc[ai][bj][i] + bv);
        }
    }
}

// ---------------------------------------------------------------------------
// Fused flash-attention: block = (b, h, 64 query rows), 4 waves x 16 rows.
// K double-buffered via async global->LDS (ASYNCcnt); V register-pipelined one
// block ahead and transposed on the LDS store. ONE barrier per key block.
// ---------------------------------------------------------------------------
__global__ __launch_bounds__(128)
void mha_attn_kernel(const _Float16* __restrict__ qh,
                     const _Float16* __restrict__ kh,
                     const _Float16* __restrict__ vh,
                     float*          __restrict__ out)
{
    __shared__ __align__(16) _Float16 lds_q[64 * 64];         // 8 KB [row][d]
    __shared__ __align__(16) _Float16 lds_k[2][32 * 64];      // 8 KB [key][d]
    __shared__ __align__(16) _Float16 lds_vt[2][64 * 32];     // 8 KB [d][key]
    __shared__ __align__(16) _Float16 lds_p[4][16 * 32];      // 4 KB wave-private P

    const int tid     = threadIdx.x;
    const int wave    = tid >> 5;
    const int lane    = tid & 31;
    const int halfsel = lane >> 4;
    const int lm      = lane & 15;

    const int q0 = blockIdx.x * 64;
    const int h  = blockIdx.y;
    const int b  = blockIdx.z;
    const size_t head_off = (size_t)(b * H_NUM + h) * S_LEN * D_HEAD;
    const _Float16* Qp = qh + head_off + (size_t)q0 * D_HEAD;
    const _Float16* Kp = kh + head_off;
    const _Float16* Vp = vh + head_off;

    // ---- Preamble: Q tile + K block 0 (async), V block 0 into registers ----
#if USE_ASYNC_COPY
#pragma unroll
    for (int i = 0; i < 4; ++i) {
        int idx = tid + i * 128;                    // 512 x 16B
        async_copy_b128(&Qp[idx * 8], &lds_q[idx * 8]);
    }
    async_copy_b128(&Kp[tid * 16],     &lds_k[0][tid * 16]);
    async_copy_b128(&Kp[tid * 16 + 8], &lds_k[0][tid * 16 + 8]);
#else
#pragma unroll
    for (int i = 0; i < 4; ++i) {
        int idx = tid + i * 128;
        *(v8h*)&lds_q[idx * 8] = *(const v8h*)&Qp[idx * 8];
    }
    *(v8h*)&lds_k[0][tid * 16]     = *(const v8h*)&Kp[tid * 16];
    *(v8h*)&lds_k[0][tid * 16 + 8] = *(const v8h*)&Kp[tid * 16 + 8];
#endif

    unsigned int vcur[8], vnxt[8];
#pragma unroll
    for (int i = 0; i < 8; ++i) {                   // V block 0: 32x64 halves
        int idx = tid + i * 128;
        int key = idx >> 5;
        int d2  = (idx & 31) << 1;
        vcur[i] = *(const unsigned int*)&Vp[(size_t)key * 64 + d2];
    }

    WAIT_ASYNC0();
    __syncthreads();

    // Q A-fragments for this wave's 16 rows (two K-slices of d).
    v16h qfr[2];
    {
        const int arow = wave * 16 + lm;
#pragma unroll
        for (int kkk = 0; kkk < 2; ++kkk) {
            int base = kkk * 32 + halfsel * 16;
            qfr[kkk] = cat8(ldsv8(&lds_q[arow * 64 + base]),
                            ldsv8(&lds_q[arow * 64 + base + 8]));
        }
    }

    v8f acc[4];
#pragma unroll
    for (int t = 0; t < 4; ++t) acc[t] = v8f_zero();
    float m_i[8], l_i[8];
#pragma unroll
    for (int i = 0; i < 8; ++i) { m_i[i] = -1e30f; l_i[i] = 0.0f; }

    for (int kb = 0; kb < S_LEN; kb += 32) {
        const int cur = (kb >> 5) & 1;
        const int nxt = cur ^ 1;

        // Store V block (registers) transposed into lds_vt[cur].
        // Buffer `cur` was last read two iterations ago (one barrier passed).
#pragma unroll
        for (int i = 0; i < 8; ++i) {
            int idx = tid + i * 128;
            int key = idx >> 5;
            int d2  = (idx & 31) << 1;
            union { unsigned int u; _Float16 hv[2]; } cv; cv.u = vcur[i];
            lds_vt[cur][(d2 + 0) * 32 + key] = cv.hv[0];
            lds_vt[cur][(d2 + 1) * 32 + key] = cv.hv[1];
        }

        WAIT_ASYNC0();      // this wave's K slice for block kb has landed in LDS
        __syncthreads();    // the only barrier: everyone's K/V visible; prev compute done

        // Prefetch block kb+32 — overlaps all the WMMAs below.
        if (kb + 32 < S_LEN) {
#if USE_ASYNC_COPY
            async_copy_b128(&Kp[(size_t)(kb + 32) * 64 + tid * 16],
                            &lds_k[nxt][tid * 16]);
            async_copy_b128(&Kp[(size_t)(kb + 32) * 64 + tid * 16 + 8],
                            &lds_k[nxt][tid * 16 + 8]);
#else
            *(v8h*)&lds_k[nxt][tid * 16] =
                *(const v8h*)&Kp[(size_t)(kb + 32) * 64 + tid * 16];
            *(v8h*)&lds_k[nxt][tid * 16 + 8] =
                *(const v8h*)&Kp[(size_t)(kb + 32) * 64 + tid * 16 + 8];
#endif
#pragma unroll
            for (int i = 0; i < 8; ++i) {
                int idx = tid + i * 128;
                int key = idx >> 5;
                int d2  = (idx & 31) << 1;
                vnxt[i] = *(const unsigned int*)&Vp[(size_t)(kb + 32 + key) * 64 + d2];
            }
        }

        // Scores: S_tile(16x32) = Q(16x64) . K^T, as two 16x16 C-frags.
        // Load all 4 K B-fragments first, then 4 back-to-back WMMAs.
        v16h kf[2][2];   // [ntile][kslice]
#pragma unroll
        for (int n = 0; n < 2; ++n)
#pragma unroll
            for (int kkk = 0; kkk < 2; ++kkk) {
                int bbase = kkk * 32 + halfsel * 16;    // contraction index = d
                kf[n][kkk] = cat8(ldsv8(&lds_k[cur][(n * 16 + lm) * 64 + bbase]),
                                  ldsv8(&lds_k[cur][(n * 16 + lm) * 64 + bbase + 8]));
            }
        v8f s0 = v8f_zero(), s1 = v8f_zero();
        s0 = wmma_f16(qfr[0], kf[0][0], s0);
        s1 = wmma_f16(qfr[0], kf[1][0], s1);
        s0 = wmma_f16(qfr[1], kf[0][1], s0);
        s1 = wmma_f16(qfr[1], kf[1][1], s1);

        // Online softmax; write P (f16) into wave-private LDS tile.
        _Float16* myp = lds_p[wave];
#pragma unroll
        for (int i = 0; i < 8; ++i) {
            float a0 = s0[i] * 0.125f;   // 1/sqrt(64)
            float a1 = s1[i] * 0.125f;
            float mx = rowmax16(fmaxf(a0, a1));
            float mn = fmaxf(m_i[i], mx);
            float corr = __expf(m_i[i] - mn);
            float p0 = __expf(a0 - mn);
            float p1 = __expf(a1 - mn);
            l_i[i] = l_i[i] * corr + rowsum16(p0 + p1);
            m_i[i] = mn;
#pragma unroll
            for (int t = 0; t < 4; ++t) acc[t][i] *= corr;
            int prow = i + halfsel * 8;
            myp[prow * 32 + lm]      = (_Float16)p0;
            myp[prow * 32 + 16 + lm] = (_Float16)p1;
        }

        // P A-fragment (16x32) — LDS ops in-order within a wave, no barrier.
        int pbase = halfsel * 16;
        v16h pf = cat8(ldsv8(&myp[lm * 32 + pbase]),
                       ldsv8(&myp[lm * 32 + pbase + 8]));

        // O(16x64) += P(16x32) . V(32x64): load 4 V B-frags, then 4 WMMAs.
        v16h vf[4];
#pragma unroll
        for (int t = 0; t < 4; ++t) {
            int vn = t * 16 + lm;
            vf[t] = cat8(ldsv8(&lds_vt[cur][vn * 32 + halfsel * 16]),
                         ldsv8(&lds_vt[cur][vn * 32 + halfsel * 16 + 8]));
        }
#pragma unroll
        for (int t = 0; t < 4; ++t)
            acc[t] = wmma_f16(pf, vf[t], acc[t]);

#pragma unroll
        for (int i = 0; i < 8; ++i) vcur[i] = vnxt[i];
    }

    // Normalize and store: out[b, q, h*64 + d], row stride = H*D = 1024.
    float* outp = out + ((size_t)b * S_LEN + q0 + wave * 16) * (H_NUM * D_HEAD)
                      + h * D_HEAD;
#pragma unroll
    for (int t = 0; t < 4; ++t) {
#pragma unroll
        for (int i = 0; i < 8; ++i) {
            int row = i + halfsel * 8;
            int col = t * 16 + lm;
            outp[(size_t)row * (H_NUM * D_HEAD) + col] = acc[t][i] / l_i[i];
        }
    }
}

// ---------------------------------------------------------------------------
extern "C" void kernel_launch(void* const* d_in, const int* in_sizes, int n_in,
                              void* d_out, int out_size, void* d_ws, size_t ws_size,
                              hipStream_t stream) {
    const float* q   = (const float*)d_in[0];
    const float* k   = (const float*)d_in[1];
    const float* v   = (const float*)d_in[2];
    const float* w_q = (const float*)d_in[3];
    const float* b_q = (const float*)d_in[4];
    const float* w_k = (const float*)d_in[5];
    const float* b_k = (const float*)d_in[6];
    const float* w_v = (const float*)d_in[7];
    const float* b_v = (const float*)d_in[8];
    float* out = (float*)d_out;

    const size_t head_elems = (size_t)B_NUM * H_NUM * S_LEN * D_HEAD; // 4M halves
    _Float16* qh = (_Float16*)d_ws;
    _Float16* kh = qh + head_elems;
    _Float16* vh = kh + head_elems;

    dim3 pg((B_NUM * S_LEN) / 64, (H_NUM * D_HEAD) / 64, 1);   // 64 x 16
    mha_proj_kernel<<<pg, 128, 0, stream>>>(q, w_q, b_q, qh);
    mha_proj_kernel<<<pg, 128, 0, stream>>>(k, w_k, b_k, kh);
    mha_proj_kernel<<<pg, 128, 0, stream>>>(v, w_v, b_v, vh);

    dim3 ag(S_LEN / 64, H_NUM, B_NUM);                          // 32 x 16 x 2
    mha_attn_kernel<<<ag, 128, 0, stream>>>(qh, kh, vh, out);
}
